// CustomSwin_63496796504591
// MI455X (gfx1250) — compile-verified, matrix operand order checked by reading, not run on hardware
//
#include <hip/hip_runtime.h>
#include <hip/hip_bf16.h>
#include <math.h>

typedef __attribute__((ext_vector_type(16))) _Float16 v16h;
typedef __attribute__((ext_vector_type(8)))  float    v8f;
typedef unsigned int v4u __attribute__((ext_vector_type(4)));
typedef int          v8i __attribute__((ext_vector_type(8)));
typedef int          v4i __attribute__((ext_vector_type(4)));

// ---------------------------------------------------------------------------
// Patch embedding: conv 4x4 stride 4 (K=48, irregular -> VALU kernel)
// ---------------------------------------------------------------------------
__global__ void k_patch_embed(const float* __restrict__ x, const float* __restrict__ w,
                              const float* __restrict__ bias, float* __restrict__ out, int B) {
    int gid = blockIdx.x * blockDim.x + threadIdx.x;
    int total = B * 4096 * 96;
    if (gid >= total) return;
    int c = gid % 96;
    int p = (gid / 96) % 4096;
    int b = gid / (96 * 4096);
    int py = p >> 6, px = p & 63;
    float acc = bias[c];
    for (int ci = 0; ci < 3; ++ci)
        for (int kh = 0; kh < 4; ++kh)
            for (int kw = 0; kw < 4; ++kw)
                acc += x[(((size_t)b * 3 + ci) * 256 + py * 4 + kh) * 256 + px * 4 + kw] *
                       w[((c * 3 + ci) * 4 + kh) * 4 + kw];
    out[gid] = acc;
}

// ---------------------------------------------------------------------------
// LayerNorm (optionally fused residual: y = res + LN(x)); wave per token
// ---------------------------------------------------------------------------
__global__ __launch_bounds__(256)
void k_layernorm(const float* __restrict__ x, const float* __restrict__ res,
                 const float* __restrict__ g, const float* __restrict__ b,
                 float* __restrict__ y, int L, int D) {
    int wave = threadIdx.x >> 5, lane = threadIdx.x & 31;
    int t = blockIdx.x * 8 + wave;
    if (t >= L) return;
    const float* xr = x + (size_t)t * D;
    float s = 0.f;
    for (int i = lane; i < D; i += 32) s += xr[i];
    for (int o = 16; o; o >>= 1) s += __shfl_xor(s, o, 32);
    float mean = s / (float)D;
    float v = 0.f;
    for (int i = lane; i < D; i += 32) { float d = xr[i] - mean; v += d * d; }
    for (int o = 16; o; o >>= 1) v += __shfl_xor(v, o, 32);
    float inv = rsqrtf(v / (float)D + 1e-5f);
    float* yr = y + (size_t)t * D;
    const float* rr = res ? res + (size_t)t * D : nullptr;
    for (int i = lane; i < D; i += 32) {
        float val = (xr[i] - mean) * inv * g[i] + b[i];
        yr[i] = rr ? rr[i] + val : val;
    }
}

// ---------------------------------------------------------------------------
// WMMA GEMM, LDS-tiled: C(M,N) = A(M,K) @ W(N,K)^T [+bias] [gelu]
//  - workgroup (8 waves) computes a 128x32 output tile
//  - A tile (128x32 f32) DMA'd into LDS by the Tensor Data Mover with
//    1-dword row padding (33-dword stride -> conflict-free fragment reads)
//  - W slice (32x32) staged once as f16 in LDS, reused by all 8 waves
//  - each wave: one A fragment feeds two v_wmma_f32_16x16x32_f16 per k-step
// Dynamic LDS only => TDM descriptor lds_addr is offset 0.
// ---------------------------------------------------------------------------
#define GEMM_LDS_BYTES (128 * 33 * 4 + 32 * 36 * 2)   // A tile (padded) + W tile

__global__ __launch_bounds__(256)
void k_gemm_wmma(const float* __restrict__ A, const float* __restrict__ Wt,
                 const float* __restrict__ bias, float* __restrict__ C,
                 int M, int N, int K, int act) {
    extern __shared__ char smem[];
    float*    As  = (float*)smem;                    // [128][33] (TDM pad)
    _Float16* Wsh = (_Float16*)(smem + 128 * 33 * 4); // [32][36]

    int lane = threadIdx.x & 31, wave = threadIdx.x >> 5;
    int half = lane >> 4, l16 = lane & 15;
    int bm = blockIdx.x * 128;
    int bn = blockIdx.y * 32;

    v8f acc0 = {}, acc1 = {};
    for (int k0 = 0; k0 < K; k0 += 32) {
        __syncthreads();   // previous k-step consumers done before overwrite
        if (wave == 0) {
            // Tensor DMA: 128 rows x 32 f32, row stride K, into LDS offset 0
            unsigned long long ga =
                (unsigned long long)(uintptr_t)(A + (size_t)bm * K + k0);
            v4u g0;
            g0[0] = 1u;                                   // count=1 (valid D#)
            g0[1] = 0u;                                   // lds_addr = 0
            g0[2] = (unsigned)(ga & 0xffffffffu);         // global_addr[31:0]
            g0[3] = (unsigned)(ga >> 32) | (2u << 30);    // addr[56:32] | type=2
            v8i g1;
            g1[0] = (2 << 16) | (1 << 20) | (4 << 22);    // 4B elems, pad_en, every 32dw +1dw
            g1[1] = (K & 0xffff) << 16;                   // tensor_dim0 lo
            g1[2] = (K >> 16) | ((M & 0xffff) << 16);     // tensor_dim0 hi | tensor_dim1 lo
            g1[3] = (M >> 16) | (32 << 16);               // tensor_dim1 hi | tile_dim0=32
            g1[4] = 128;                                  // tile_dim1=128
            g1[5] = K;                                    // tensor_dim0_stride lo
            g1[6] = 0;
            g1[7] = 0;
            v4i gz = {0, 0, 0, 0};
#if __clang_major__ >= 23
            v8i gz8 = {0, 0, 0, 0, 0, 0, 0, 0};
            __builtin_amdgcn_tensor_load_to_lds(g0, g1, gz, gz, gz8, 0);
#else
            __builtin_amdgcn_tensor_load_to_lds(g0, g1, gz, gz, 0);
#endif
        }
        // cooperatively stage W slice as f16 (overlaps with the TDM transfer)
        for (int idx = threadIdx.x; idx < 1024; idx += 256) {
            int n = idx >> 5, kk = idx & 31;
            __builtin_prefetch(&Wt[(size_t)(bn + n) * K + k0 + 32], 0, 1);
            Wsh[n * 36 + kk] = (_Float16)Wt[(size_t)(bn + n) * K + k0 + kk];
        }
        if (wave == 0) __builtin_amdgcn_s_wait_tensorcnt(0);
        __syncthreads();

        const float* arow = As + (wave * 16 + l16) * 33;
        v16h a, b0, b1;
#pragma unroll
        for (int vz = 0; vz < 8; ++vz) {
            int kk = half * 8 + ((vz & 4) << 2) + ((vz & 3) << 1);
            a[2 * vz]      = (_Float16)arow[kk];
            a[2 * vz + 1]  = (_Float16)arow[kk + 1];
            b0[2 * vz]     = Wsh[l16 * 36 + kk];
            b0[2 * vz + 1] = Wsh[l16 * 36 + kk + 1];
            b1[2 * vz]     = Wsh[(16 + l16) * 36 + kk];
            b1[2 * vz + 1] = Wsh[(16 + l16) * 36 + kk + 1];
        }
        acc0 = __builtin_amdgcn_wmma_f32_16x16x32_f16(false, a, false, b0,
                                                      (short)0, acc0, false, false);
        acc1 = __builtin_amdgcn_wmma_f32_16x16x32_f16(false, a, false, b1,
                                                      (short)0, acc1, false, false);
    }
#pragma unroll
    for (int j = 0; j < 8; ++j) {
        int row  = bm + wave * 16 + half * 8 + j;
        int col0 = bn + l16;
        int col1 = bn + 16 + l16;
        float v0 = acc0[j], v1 = acc1[j];
        if (bias) { v0 += bias[col0]; v1 += bias[col1]; }
        if (act == 1) {
            v0 = v0 * 0.5f * (1.0f + erff(v0 * 0.70710678118f));
            v1 = v1 * 0.5f * (1.0f + erff(v1 * 0.70710678118f));
        }
        C[(size_t)row * N + col0] = v0;
        C[(size_t)row * N + col1] = v1;
    }
}

// ---------------------------------------------------------------------------
// CPB MLP: table(225,2) -> relu(512) -> heads; raw logits (sigmoid in attn)
// ---------------------------------------------------------------------------
__global__ __launch_bounds__(256)
void k_cpb(const float* __restrict__ w1, const float* __restrict__ b1,
           const float* __restrict__ w2, float* __restrict__ tab, int heads) {
    __shared__ float hid[512];
    int i = blockIdx.x;                // 0..224
    int dy = i / 15 - 7, dx = i % 15 - 7;
    float ty = (float)dy / 7.f * 8.f;
    float tx = (float)dx / 7.f * 8.f;
    float fy = (ty >= 0.f ? 1.f : -1.f) * log2f(fabsf(ty) + 1.f) * (1.f / 3.f);
    float fx = (tx >= 0.f ? 1.f : -1.f) * log2f(fabsf(tx) + 1.f) * (1.f / 3.f);
    for (int j = threadIdx.x; j < 512; j += 256)
        hid[j] = fmaxf(fy * w1[2 * j] + fx * w1[2 * j + 1] + b1[j], 0.f);
    __syncthreads();
    for (int h = threadIdx.x; h < heads; h += 256) {
        float acc = 0.f;
        for (int j = 0; j < 512; ++j) acc += hid[j] * w2[h * 512 + j];
        tab[h * 225 + i] = acc;
    }
}

// ---------------------------------------------------------------------------
// Window attention (SwinV2 cosine attention), one workgroup per (b, window, head)
// head_dim == 32 always; window = 8x8 = 64 tokens. WMMA for S=qk^T and O=PV.
// ---------------------------------------------------------------------------
__device__ __forceinline__ int regid(int pos, int H, int shift) {
    return pos < H - 8 ? 0 : (pos < H - shift ? 1 : 2);
}

__global__ __launch_bounds__(256)
void k_win_attn(const float* __restrict__ Q, const float* __restrict__ K,
                const float* __restrict__ V, const float* __restrict__ lscale,
                const float* __restrict__ cpb, float* __restrict__ Out,
                int Himg, int dim, int heads, int shift) {
    __shared__ float qs[64][32];
    __shared__ float ks[64][32];
    __shared__ float vs[64][32];
    __shared__ float S[64][64];
    int nWside = Himg >> 3;
    int nW = nWside * nWside;
    int b = blockIdx.x / nW;
    int w = blockIdx.x % nW;
    int head = blockIdx.y;
    int wy = w / nWside, wx = w % nWside;
    int tid = threadIdx.x;
    int lane = tid & 31, wave = tid >> 5;
    int half = lane >> 4, l16 = lane & 15;

    // gather Q,K,V (shifted windows: roll(-shift) == +shift index)
    for (int i = tid; i < 64 * 32; i += 256) {
        int e = i >> 5, d = i & 31;
        int r = e >> 3, c = e & 7;
        int hy = (wy * 8 + r + shift) % Himg;
        int hx = (wx * 8 + c + shift) % Himg;
        size_t gi = ((size_t)b * Himg * Himg + hy * Himg + hx) * dim + head * 32 + d;
        qs[e][d] = Q[gi]; ks[e][d] = K[gi]; vs[e][d] = V[gi];
    }
    __syncthreads();

    // cosine normalize rows of q and k (head_dim 32 == wave width)
    for (int e = wave; e < 64; e += 8) {
        float qv = qs[e][lane], kv = ks[e][lane];
        float sq = qv * qv, sk = kv * kv;
        for (int o = 16; o; o >>= 1) { sq += __shfl_xor(sq, o, 32); sk += __shfl_xor(sk, o, 32); }
        qs[e][lane] = qv / fmaxf(sqrtf(sq), 1e-12f);
        ks[e][lane] = kv / fmaxf(sqrtf(sk), 1e-12f);
    }
    __syncthreads();

    float scale = __expf(fminf(lscale[head], 4.60517019f)); // log(100)

    // S = qn @ kn^T (16 tiles of 16x16, K=32 -> one WMMA each)
    for (int t = wave; t < 16; t += 8) {
        int tm = t >> 2, tn = t & 3;
        v16h a, bm;
#pragma unroll
        for (int vz = 0; vz < 8; ++vz) {
            int kk = half * 8 + ((vz & 4) << 2) + ((vz & 3) << 1);
            a[2 * vz]      = (_Float16)qs[tm * 16 + l16][kk];
            a[2 * vz + 1]  = (_Float16)qs[tm * 16 + l16][kk + 1];
            bm[2 * vz]     = (_Float16)ks[tn * 16 + l16][kk];
            bm[2 * vz + 1] = (_Float16)ks[tn * 16 + l16][kk + 1];
        }
        v8f acc = {};
        acc = __builtin_amdgcn_wmma_f32_16x16x32_f16(false, a, false, bm,
                                                     (short)0, acc, false, false);
#pragma unroll
        for (int j = 0; j < 8; ++j) {
            int rn = tm * 16 + half * 8 + j;   // query element in window
            int cn = tn * 16 + l16;            // key element in window
            int r1 = rn >> 3, c1 = rn & 7, r2 = cn >> 3, c2 = cn & 7;
            int ridx = (r1 - r2 + 7) * 15 + (c1 - c2 + 7);
            float bias = 16.f / (1.f + __expf(-cpb[head * 225 + ridx]));
            float val = acc[j] * scale + bias;
            if (shift) {
                int g1 = 3 * regid(wy * 8 + r1, Himg, shift) + regid(wx * 8 + c1, Himg, shift);
                int g2 = 3 * regid(wy * 8 + r2, Himg, shift) + regid(wx * 8 + c2, Himg, shift);
                if (g1 != g2) val -= 100.f;
            }
            S[rn][cn] = val;
        }
    }
    __syncthreads();

    // row softmax over 64 keys (2 elems/lane)
    for (int it = 0; it < 8; ++it) {
        int rn = wave * 8 + it;
        float v0 = S[rn][lane], v1 = S[rn][lane + 32];
        float mx = fmaxf(v0, v1);
        for (int o = 16; o; o >>= 1) mx = fmaxf(mx, __shfl_xor(mx, o, 32));
        v0 = __expf(v0 - mx); v1 = __expf(v1 - mx);
        float sm = v0 + v1;
        for (int o = 16; o; o >>= 1) sm += __shfl_xor(sm, o, 32);
        float inv = 1.f / sm;
        S[rn][lane] = v0 * inv; S[rn][lane + 32] = v1 * inv;
    }
    __syncthreads();

    // O = P @ V : 8 tiles (4x2), one per wave, K=64 -> two WMMA steps
    {
        int tm = wave >> 1, tn = wave & 1;
        v8f acc = {};
        for (int k0 = 0; k0 < 64; k0 += 32) {
            v16h a, bm;
#pragma unroll
            for (int vz = 0; vz < 8; ++vz) {
                int kk = k0 + half * 8 + ((vz & 4) << 2) + ((vz & 3) << 1);
                a[2 * vz]      = (_Float16)S[tm * 16 + l16][kk];
                a[2 * vz + 1]  = (_Float16)S[tm * 16 + l16][kk + 1];
                bm[2 * vz]     = (_Float16)vs[kk][tn * 16 + l16];
                bm[2 * vz + 1] = (_Float16)vs[kk + 1][tn * 16 + l16];
            }
            acc = __builtin_amdgcn_wmma_f32_16x16x32_f16(false, a, false, bm,
                                                         (short)0, acc, false, false);
        }
#pragma unroll
        for (int j = 0; j < 8; ++j) {
            int e = tm * 16 + half * 8 + j;
            int d = tn * 16 + l16;
            int r = e >> 3, c = e & 7;
            int hy = (wy * 8 + r + shift) % Himg;
            int hx = (wx * 8 + c + shift) % Himg;
            size_t gi = ((size_t)b * Himg * Himg + hy * Himg + hx) * dim + head * 32 + d;
            Out[gi] = acc[j];
        }
    }
}

// ---------------------------------------------------------------------------
// Patch merging gather: (B,H,W,C) -> (B, H/2*W/2, 4C), chunk order per reference
// ---------------------------------------------------------------------------
__global__ void k_merge(const float* __restrict__ h, float* __restrict__ cat,
                        int B, int Himg, int dim) {
    int H2 = Himg >> 1;
    int total = B * H2 * H2 * 4 * dim;
    int gid = blockIdx.x * blockDim.x + threadIdx.x;
    if (gid >= total) return;
    int c4 = gid % (4 * dim);
    int p  = (gid / (4 * dim)) % (H2 * H2);
    int b  = gid / (4 * dim * H2 * H2);
    int qd = c4 / dim, c = c4 % dim;
    int dy = qd & 1, dx = qd >> 1;   // chunks: (0,0),(1,0),(0,1),(1,1)
    int y2 = p / H2, x2 = p % H2;
    cat[gid] = h[(((size_t)b * Himg + (2 * y2 + dy)) * Himg + (2 * x2 + dx)) * dim + c];
}

// channel-mean pool: out[token] = mean_c h[token, c]
__global__ void k_chanmean(const float* __restrict__ h, float* __restrict__ out,
                           int total, int D) {
    int gid = blockIdx.x * blockDim.x + threadIdx.x;
    if (gid >= total) return;
    const float* r = h + (size_t)gid * D;
    float s = 0.f;
    for (int c = 0; c < D; ++c) s += r[c];
    out[gid] = s / (float)D;
}

// token-mean: pooled[b,k] = mean_t seq[b,t,k]  (T=64, D=768)
__global__ void k_tokenmean(const float* __restrict__ seq, float* __restrict__ pooled) {
    int gid = blockIdx.x * blockDim.x + threadIdx.x;
    if (gid >= 8 * 768) return;
    int b = gid / 768, k = gid % 768;
    float s = 0.f;
    for (int t = 0; t < 64; ++t) s += seq[((size_t)b * 64 + t) * 768 + k];
    pooled[gid] = s * (1.f / 64.f);
}

// classifier head: out[b,o] = pooled[b,:] . head_w[o,:] + head_b[o]
__global__ void k_head(const float* __restrict__ pooled, const float* __restrict__ hw,
                       const float* __restrict__ hb, float* __restrict__ out) {
    int gid = blockIdx.x * blockDim.x + threadIdx.x;
    if (gid >= 8 * 512) return;
    int b = gid / 512, o = gid % 512;
    float acc = hb[o];
    for (int k = 0; k < 768; ++k) acc += pooled[b * 768 + k] * hw[o * 768 + k];
    out[gid] = acc;
}

// ---------------------------------------------------------------------------
// Host-side orchestration
// ---------------------------------------------------------------------------
struct Blk {
    const float *wq, *bq, *wk, *wv, *bv, *ls, *c1w, *c1b, *c2w, *pw, *pb;
    const float *l1g, *l1b, *f1w, *f1b, *f2w, *f2b, *l2g, *l2b;
};

static const int DEPTHS[4] = {2, 2, 6, 2};
static const int HEADSARR[4] = {3, 6, 12, 24};

extern "C" void kernel_launch(void* const* d_in, const int* in_sizes, int n_in,
                              void* d_out, int out_size, void* d_ws, size_t ws_size,
                              hipStream_t stream) {
    (void)in_sizes; (void)n_in; (void)out_size; (void)ws_size;
    int idx = 0;
    auto next = [&]() -> const float* { return (const float*)d_in[idx++]; };

    // setup_inputs() insertion order: x, then params (insertion order)
    const float* x        = next();
    const float* patch_w  = next();
    const float* patch_b  = next();
    const float* patch_lg = next();
    const float* patch_lb = next();
    Blk blk[4][6];
    const float *red_w[3], *ds_g[3], *ds_b[3];
    for (int si = 0; si < 4; ++si) {
        for (int bi = 0; bi < DEPTHS[si]; ++bi) {
            Blk& p = blk[si][bi];
            p.wq = next(); p.bq = next(); p.wk = next(); p.wv = next(); p.bv = next();
            p.ls = next(); p.c1w = next(); p.c1b = next(); p.c2w = next();
            p.pw = next(); p.pb = next(); p.l1g = next(); p.l1b = next();
            p.f1w = next(); p.f1b = next(); p.f2w = next(); p.f2b = next();
            p.l2g = next(); p.l2b = next();
        }
        if (si < 3) { red_w[si] = next(); ds_g[si] = next(); ds_b[si] = next(); }
    }
    const float* fin_g  = next();
    const float* fin_b  = next();
    const float* head_w = next();
    const float* head_b = next();

    float* out = (float*)d_out;
    float* wsf = (float*)d_ws;
    // workspace layout (floats): ~22.6M floats (~90 MB)
    float* h      = wsf + 0;          // <= 3,145,728
    float* t1     = wsf + 3200000;    // LN / attention out
    float* t2     = wsf + 6400000;    // proj out / mlp out / merge cat
    float* big    = wsf + 9600000;    // mlp hidden (<=12.58M); aliases q/k/v
    float* qb     = big;
    float* kb     = wsf + 12800000;
    float* vb     = wsf + 16000000;
    float* cpbbuf = wsf + 22400000;   // heads*225 <= 5400
    float* pooled = wsf + 22450000;   // 8*768

    const int B = 8;
    // d_out layout: head [0,4096), pools at 4096 / 36864 / 45056 / 47104
    const int poolOff[4] = {4096, 36864, 45056, 47104};

    // ---- patch embed + LN + pool0 ----
    k_patch_embed<<<(B * 4096 * 96 + 255) / 256, 256, 0, stream>>>(x, patch_w, patch_b, t1, B);
    k_layernorm<<<(B * 4096 + 7) / 8, 256, 0, stream>>>(t1, nullptr, patch_lg, patch_lb, h, B * 4096, 96);
    k_chanmean<<<(B * 4096 + 255) / 256, 256, 0, stream>>>(h, out + poolOff[0], B * 4096, 96);

    int Himg = 64, dim = 96;
    for (int si = 0; si < 4; ++si) {
        int heads = HEADSARR[si];
        int M = B * Himg * Himg;
        for (int bi = 0; bi < DEPTHS[si]; ++bi) {
            const Blk& p = blk[si][bi];
            int shift = ((bi & 1) && Himg > 8) ? 4 : 0;
            k_cpb<<<225, 256, 0, stream>>>(p.c1w, p.c1b, p.c2w, cpbbuf, heads);
            // QKV projections (per-token linear; window permutation handled in attn)
            k_gemm_wmma<<<dim3(M / 128, dim / 32), 256, GEMM_LDS_BYTES, stream>>>(
                h, p.wq, p.bq, qb, M, dim, dim, 0);
            k_gemm_wmma<<<dim3(M / 128, dim / 32), 256, GEMM_LDS_BYTES, stream>>>(
                h, p.wk, nullptr, kb, M, dim, dim, 0);
            k_gemm_wmma<<<dim3(M / 128, dim / 32), 256, GEMM_LDS_BYTES, stream>>>(
                h, p.wv, p.bv, vb, M, dim, dim, 0);
            int nW = (Himg / 8) * (Himg / 8);
            dim3 ag(B * nW, heads);
            k_win_attn<<<ag, 256, 0, stream>>>(qb, kb, vb, p.ls, cpbbuf, t1, Himg, dim, heads, shift);
            k_gemm_wmma<<<dim3(M / 128, dim / 32), 256, GEMM_LDS_BYTES, stream>>>(
                t1, p.pw, p.pb, t2, M, dim, dim, 0);
            // res-post-norm: h = h + LN(attn_proj)
            k_layernorm<<<(M + 7) / 8, 256, 0, stream>>>(t2, h, p.l1g, p.l1b, h, M, dim);
            // MLP with exact GELU
            k_gemm_wmma<<<dim3(M / 128, (4 * dim) / 32), 256, GEMM_LDS_BYTES, stream>>>(
                h, p.f1w, p.f1b, big, M, 4 * dim, dim, 1);
            k_gemm_wmma<<<dim3(M / 128, dim / 32), 256, GEMM_LDS_BYTES, stream>>>(
                big, p.f2w, p.f2b, t2, M, dim, 4 * dim, 0);
            k_layernorm<<<(M + 7) / 8, 256, 0, stream>>>(t2, h, p.l2g, p.l2b, h, M, dim);
        }
        if (si < 3) {
            // patch merging: gather 2x2 -> Linear(4C->2C) -> LN
            int catTotal = B * (Himg / 2) * (Himg / 2) * 4 * dim;
            k_merge<<<(catTotal + 255) / 256, 256, 0, stream>>>(h, t2, B, Himg, dim);
            Himg >>= 1;
            int M2 = B * Himg * Himg;
            k_gemm_wmma<<<dim3(M2 / 128, (2 * dim) / 32), 256, GEMM_LDS_BYTES, stream>>>(
                t2, red_w[si], nullptr, t1, M2, 2 * dim, 4 * dim, 0);
            dim *= 2;
            k_layernorm<<<(M2 + 7) / 8, 256, 0, stream>>>(t1, nullptr, ds_g[si], ds_b[si], h, M2, dim);
            k_chanmean<<<(M2 + 255) / 256, 256, 0, stream>>>(h, out + poolOff[si + 1], M2, dim);
        }
    }

    // ---- final LN, token-mean pool, head ----
    k_layernorm<<<(B * 64 + 7) / 8, 256, 0, stream>>>(h, nullptr, fin_g, fin_b, t1, B * 64, 768);
    k_tokenmean<<<(8 * 768 + 255) / 256, 256, 0, stream>>>(t1, pooled);
    k_head<<<(8 * 512 + 255) / 256, 256, 0, stream>>>(pooled, head_w, head_b, out);
}
